// GraphRNNCell_26113401160016
// MI455X (gfx1250) — matrix operand order repaired and checked by reading, not compiled, not run on hardware
//
#include <hip/hip_runtime.h>
#include <hip/hip_bf16.h>

// ---------------- problem constants ----------------
#define BB   8
#define NN   10000
#define EE   160000
#define BN   80000          // BB*NN
#define DH   128
#define KD   256            // D_IN + D_H
#define EPSV 1e-5f

typedef __attribute__((ext_vector_type(16))) __bf16 v16bf;
typedef __attribute__((ext_vector_type(8)))  float  v8f;

// A-fragment K offset inside a 32-wide chunk (16-bit A, 16x32, ISA 7.12.2)
__device__ __forceinline__ int koffA(int e, int g) {
    return (e < 8) ? (8 * g + e) : (8 + 8 * g + e);   // e>=8 -> 16 + 8g + (e-8)
}

// ---------------- degree / norm ----------------
__global__ __launch_bounds__(256) void k_fill1(float* __restrict__ p, int n) {
    int i = blockIdx.x * 256 + threadIdx.x;
    if (i < n) p[i] = 1.0f;                           // self-loop contributes 1 to degree
}

__global__ __launch_bounds__(256) void k_degree(const int* __restrict__ ei,
                                                float* __restrict__ deg) {
    int idx = blockIdx.x * 256 + threadIdx.x;         // grid exact: BB*EE threads
    int b = idx / EE;
    int e = idx - b * EE;
    int d = ei[EE + e] + b * NN;
    atomicAdd(deg + d, 1.0f);
}

__global__ __launch_bounds__(256) void k_rsqrt(float* __restrict__ p, int n) {
    int i = blockIdx.x * 256 + threadIdx.x;
    if (i < n) p[i] = rsqrtf(p[i]);                   // deg >= 1 always (self loops)
}

// ---------------- WMMA GEMM:  T[80000x128] = concat(X0,X1)[80000x256] @ W[256x128] ----
// 256 threads = 8 waves; block computes 128 rows x 128 cols.
// bf16 hi/lo decomposition: a*w ~= ah*wh + al*wh + ah*wl   (3 WMMAs / tile / chunk)
__global__ __launch_bounds__(256) void k_gemm(const float* __restrict__ X0,
                                              const float* __restrict__ X1,
                                              const float* __restrict__ W,
                                              float* __restrict__ T) {
    __shared__ __bf16 Ahi[128 * 32];
    __shared__ __bf16 Alo[128 * 32];

    const int t    = threadIdx.x;
    const int lane = t & 31;
    const int wv   = t >> 5;            // 0..7 : this wave's 16-column tile
    const int g    = lane >> 4;
    const int ln   = lane & 15;
    const int row0 = blockIdx.x * 128;  // 625 blocks, 80000 = 625*128 exact
    const int ncol = wv * 16 + ln;

    v8f acc[8];
#pragma unroll
    for (int mt = 0; mt < 8; ++mt)
#pragma unroll
        for (int i = 0; i < 8; ++i) acc[mt][i] = 0.0f;

    for (int c = 0; c < 8; ++c) {       // K chunks of 32
        __syncthreads();
        // ---- stage A chunk (128 rows x 32 k) as bf16 hi/lo in LDS, coalesced ----
#pragma unroll
        for (int i = 0; i < 16; ++i) {
            int idx = t + i * 256;                    // 4096 elements
            int r   = idx >> 5;
            int kk  = idx & 31;
            int kg  = c * 32 + kk;
            float v = (kg < 128) ? X0[(row0 + r) * 128 + kg]
                                 : X1[(row0 + r) * 128 + (kg - 128)];
            __bf16 hi = (__bf16)v;
            Ahi[idx] = hi;
            Alo[idx] = (__bf16)(v - (float)hi);
        }
        __syncthreads();

        // ---- B fragment (32x16 slice of W) straight from global (L2-resident) ----
        v16bf bhi, blo;
#pragma unroll
        for (int e = 0; e < 16; ++e) {                // B layout: K = 16*g + e
            int k   = c * 32 + 16 * g + e;
            float w = W[k * DH + ncol];
            __bf16 hi = (__bf16)w;
            bhi[e] = hi;
            blo[e] = (__bf16)(w - (float)hi);
        }

#pragma unroll
        for (int mt = 0; mt < 8; ++mt) {
            v16bf ahi, alo;
            const int rb = (mt * 16 + ln) * 32;
#pragma unroll
            for (int e = 0; e < 16; ++e) {
                int kk = 16 * g + e;
                ahi[e] = Ahi[rb + kk];
                alo[e] = Alo[rb + kk];
            }
            acc[mt] = __builtin_amdgcn_wmma_f32_16x16x32_bf16(
                false, ahi, false, bhi, (short)0, acc[mt], false, false);
            acc[mt] = __builtin_amdgcn_wmma_f32_16x16x32_bf16(
                false, alo, false, bhi, (short)0, acc[mt], false, false);
            acc[mt] = __builtin_amdgcn_wmma_f32_16x16x32_bf16(
                false, ahi, false, blo, (short)0, acc[mt], false, false);
        }
    }

    // ---- store D tiles (C/D layout: VGPR i -> M = 8*g + i, N = ln) ----
#pragma unroll
    for (int mt = 0; mt < 8; ++mt)
#pragma unroll
        for (int i = 0; i < 8; ++i) {
            int r = row0 + mt * 16 + g * 8 + i;
            T[r * DH + ncol] = acc[mt][i];
        }
}

// ---------------- self-loop init:  A = T * dis[i]^2 ----------------
__global__ __launch_bounds__(256) void k_selfinit2(const float* __restrict__ Tz,
                                                   const float* __restrict__ Tr,
                                                   const float* __restrict__ dis,
                                                   float* __restrict__ Az,
                                                   float* __restrict__ Ar) {
    int idx  = blockIdx.x * 256 + threadIdx.x;        // grid exact: BN*DH threads
    int node = idx >> 7;
    float d  = dis[node];
    float s2 = d * d;
    Az[idx] = Tz[idx] * s2;
    Ar[idx] = Tr[idx] * s2;
}

__global__ __launch_bounds__(256) void k_selfinit1(const float* __restrict__ Tc,
                                                   const float* __restrict__ dis,
                                                   float* __restrict__ Ac) {
    int idx  = blockIdx.x * 256 + threadIdx.x;
    int node = idx >> 7;
    float d  = dis[node];
    Ac[idx] = Tc[idx] * d * d;
}

// ---------------- edge scatter: one wave per edge, lane handles 4 feats ----------------
__global__ __launch_bounds__(256) void k_scatter2(const int* __restrict__ ei,
                                                  const float* __restrict__ dis,
                                                  const float* __restrict__ Tz,
                                                  const float* __restrict__ Tr,
                                                  float* __restrict__ Az,
                                                  float* __restrict__ Ar) {
    unsigned idx = blockIdx.x * 256u + threadIdx.x;   // grid exact: BB*EE*32 threads
    int ge = (int)(idx >> 5);                         // edge id in [0, BB*EE)
    int l  = (int)(idx & 31);
    int b  = ge / EE;
    int e  = ge - b * EE;
    int s  = ei[e] + b * NN;
    int d  = ei[EE + e] + b * NN;
    float nrm = dis[s] * dis[d];

    const float4 vz = ((const float4*)(Tz + (size_t)s * DH))[l];
    const float4 vr = ((const float4*)(Tr + (size_t)s * DH))[l];
    float* az = Az + (size_t)d * DH + l * 4;
    float* ar = Ar + (size_t)d * DH + l * 4;
    atomicAdd(az + 0, vz.x * nrm);
    atomicAdd(az + 1, vz.y * nrm);
    atomicAdd(az + 2, vz.z * nrm);
    atomicAdd(az + 3, vz.w * nrm);
    atomicAdd(ar + 0, vr.x * nrm);
    atomicAdd(ar + 1, vr.y * nrm);
    atomicAdd(ar + 2, vr.z * nrm);
    atomicAdd(ar + 3, vr.w * nrm);
}

__global__ __launch_bounds__(256) void k_scatter1(const int* __restrict__ ei,
                                                  const float* __restrict__ dis,
                                                  const float* __restrict__ Tc,
                                                  float* __restrict__ Ac) {
    unsigned idx = blockIdx.x * 256u + threadIdx.x;
    int ge = (int)(idx >> 5);
    int l  = (int)(idx & 31);
    int b  = ge / EE;
    int e  = ge - b * EE;
    int s  = ei[e] + b * NN;
    int d  = ei[EE + e] + b * NN;
    float nrm = dis[s] * dis[d];

    const float4 vc = ((const float4*)(Tc + (size_t)s * DH))[l];
    float* ac = Ac + (size_t)d * DH + l * 4;
    atomicAdd(ac + 0, vc.x * nrm);
    atomicAdd(ac + 1, vc.y * nrm);
    atomicAdd(ac + 2, vc.z * nrm);
    atomicAdd(ac + 3, vc.w * nrm);
}

// ---------------- gate: z=sig(Az+bz) in-place; rh = sig(Ar+br)*h in-place ----------------
__device__ __forceinline__ float sigm(float x) { return 1.0f / (1.0f + __expf(-x)); }

__global__ __launch_bounds__(256) void k_gate(float* __restrict__ Az,       // -> z
                                              float* __restrict__ Ar,       // -> r*h
                                              const float* __restrict__ bz,
                                              const float* __restrict__ br,
                                              const float* __restrict__ H) {
    int idx = blockIdx.x * 256 + threadIdx.x;         // grid exact: BN*DH
    int dc  = idx & (DH - 1);
    float z = sigm(Az[idx] + bz[dc]);
    float r = sigm(Ar[idx] + br[dc]);
    Az[idx] = z;
    Ar[idx] = r * H[idx];
}

// ---------------- final: GRU combine + LayerNorm (one wave32 per node) ----------------
__global__ __launch_bounds__(256) void k_final(const float* __restrict__ Ac,
                                               const float* __restrict__ Z,
                                               const float* __restrict__ H,
                                               const float* __restrict__ bc,
                                               const float* __restrict__ gamma,
                                               const float* __restrict__ beta,
                                               float* __restrict__ out) {
    const int lane = threadIdx.x & 31;
    const int wv   = threadIdx.x >> 5;
    const int node = blockIdx.x * 8 + wv;             // grid exact: BN/8 blocks
    const int base = node * DH + lane * 4;

    float hn[4];
#pragma unroll
    for (int j = 0; j < 4; ++j) {
        float z  = Z[base + j];
        float hc = tanhf(Ac[base + j] + bc[lane * 4 + j]);
        hn[j] = (1.0f - z) * H[base + j] + z * hc;
    }
    float s = hn[0] + hn[1] + hn[2] + hn[3];
#pragma unroll
    for (int off = 16; off > 0; off >>= 1) s += __shfl_xor(s, off, 32);
    float mu = s * (1.0f / 128.0f);

    float v = 0.0f;
#pragma unroll
    for (int j = 0; j < 4; ++j) { float d = hn[j] - mu; v += d * d; }
#pragma unroll
    for (int off = 16; off > 0; off >>= 1) v += __shfl_xor(v, off, 32);
    float inv = rsqrtf(v * (1.0f / 128.0f) + EPSV);

#pragma unroll
    for (int j = 0; j < 4; ++j)
        out[base + j] = (hn[j] - mu) * inv * gamma[lane * 4 + j] + beta[lane * 4 + j];
}

// ---------------- host launcher ----------------
extern "C" void kernel_launch(void* const* d_in, const int* in_sizes, int n_in,
                              void* d_out, int out_size, void* d_ws, size_t ws_size,
                              hipStream_t stream) {
    const float* x     = (const float*)d_in[0];
    const int*   ei    = (const int*)  d_in[1];
    const float* h     = (const float*)d_in[2];
    const float* Wz    = (const float*)d_in[3];
    const float* bz    = (const float*)d_in[4];
    const float* Wr    = (const float*)d_in[5];
    const float* br    = (const float*)d_in[6];
    const float* Wc    = (const float*)d_in[7];
    const float* bc    = (const float*)d_in[8];
    const float* gamma = (const float*)d_in[9];
    const float* beta  = (const float*)d_in[10];
    float*       out   = (float*)d_out;

    // workspace carve-out (~165 MB): dis + 4 node-feature buffers
    float* dis  = (float*)d_ws;                 // BN floats
    float* buf0 = dis  + BN;                    // Tz, later Tc
    float* buf1 = buf0 + (size_t)BN * DH;       // Tr, later Ac
    float* buf2 = buf1 + (size_t)BN * DH;       // Az, later z
    float* buf3 = buf2 + (size_t)BN * DH;       // Ar, later r*h

    const int nodeBlocks = (BN + 255) / 256;
    const int featBlocks = (BN * DH) / 256;     // 40000, exact
    const int edgeBlocks = (BB * EE) / 256;     // 5000, exact
    const int scatBlocks = (BB * EE * 32) / 256;// 160000, exact
    const int gemmBlocks = BN / 128;            // 625, exact

    // 1) symmetric normalization: deg (with self loop) -> 1/sqrt(deg)
    k_fill1 <<<nodeBlocks, 256, 0, stream>>>(dis, BN);
    k_degree<<<edgeBlocks, 256, 0, stream>>>(ei, dis);
    k_rsqrt <<<nodeBlocks, 256, 0, stream>>>(dis, BN);

    // 2) z / r linear transforms (shared A = concat(x, h))
    k_gemm<<<gemmBlocks, 256, 0, stream>>>(x, h, Wz, buf0);
    k_gemm<<<gemmBlocks, 256, 0, stream>>>(x, h, Wr, buf1);

    // 3) propagate: self-loop term + fused z/r edge scatter
    k_selfinit2<<<featBlocks, 256, 0, stream>>>(buf0, buf1, dis, buf2, buf3);
    k_scatter2 <<<scatBlocks, 256, 0, stream>>>(ei, dis, buf0, buf1, buf2, buf3);

    // 4) gates: buf2 <- z, buf3 <- r*h
    k_gate<<<featBlocks, 256, 0, stream>>>(buf2, buf3, bz, br, h);

    // 5) candidate conv on concat(x, r*h)
    k_gemm     <<<gemmBlocks, 256, 0, stream>>>(x, buf3, Wc, buf0);
    k_selfinit1<<<featBlocks, 256, 0, stream>>>(buf0, dis, buf1);
    k_scatter1 <<<scatBlocks, 256, 0, stream>>>(ei, dis, buf0, buf1);

    // 6) GRU combine + LayerNorm
    k_final<<<BN / 8, 256, 0, stream>>>(buf1, buf2, h, bc, gamma, beta, out);

    (void)in_sizes; (void)n_in; (void)out_size; (void)ws_size;
}
